// RSSM_86792699117824
// MI455X (gfx1250) — matrix operand order repaired and checked by reading, not compiled, not run on hardware
//
#include <hip/hip_runtime.h>
#include <hip/hip_bf16.h>
#include <stdint.h>

// ---------------- problem dims (fixed by the reference) ----------------
static constexpr int T   = 64;
static constexpr int B   = 512;
static constexpr int D   = 1024;   // deterministic / belief width
static constexpr int Z   = 256;    // stochastic state width
static constexpr int ACT = 32;     // action width
static constexpr int H   = 512;    // hidden width
static constexpr int E   = 1024;   // observation embedding width
static constexpr int KSA = Z + ACT; // 288

typedef __attribute__((ext_vector_type(16))) __bf16 v16bf;
typedef __attribute__((ext_vector_type(8)))  float  v8f;

// ---------------- small helpers ----------------
__device__ __forceinline__ unsigned short f2bf(float f) {
  union { float f; unsigned u; } x{f};
  unsigned u = x.u + 0x7fffu + ((x.u >> 16) & 1u);   // round-to-nearest-even
  return (unsigned short)(u >> 16);
}
__device__ __forceinline__ float sigm(float x)  { return 1.f / (1.f + __expf(-x)); }
__device__ __forceinline__ float softp(float x) { return log1pf(__expf(-fabsf(x))) + fmaxf(x, 0.f); }

union FragB { v16bf v; uint4 q[2]; unsigned short s[16]; };

__device__ __forceinline__ v8f wmma_bf16(v16bf a, v16bf b, v8f c) {
  // D = A(16x32 bf16) * B(32x16 bf16) + C(16x16 f32)
  return __builtin_amdgcn_wmma_f32_16x16x32_bf16(false, a, false, b, (short)0, c, false, false);
}

// A fragment (16x32 row-major tile) from LDS bf16; rows 16B-aligned (ld % 8 == 0).
// ISA layout: lanes 0-15 -> M, VGPR i<4 -> K = 2i + 8*half, i>=4 -> K = 16 + 2(i-4) + 8*half.
__device__ __forceinline__ v16bf load_a_lds(const unsigned short* Xs, int ld, int kb) {
  const int lane = threadIdx.x & 31;
  const int m = lane & 15, hf = lane >> 4;
  FragB f;
  const unsigned short* row = Xs + m * ld + kb + 8 * hf;
  f.q[0] = *(const uint4*)(row);        // K = kb + 8*hf + 0..7
  f.q[1] = *(const uint4*)(row + 16);   // K = kb + 8*hf + 16..23
  return f.v;
}
// Same A layout but straight from a global row-major bf16 matrix (rows b0..b0+15).
__device__ __forceinline__ v16bf load_a_gbl(const unsigned short* X, int ld, int b0, int kb) {
  const int lane = threadIdx.x & 31;
  const int m = lane & 15, hf = lane >> 4;
  FragB f;
  const unsigned short* row = X + (size_t)(b0 + m) * ld + kb + 8 * hf;
  f.q[0] = *(const uint4*)(row);
  f.q[1] = *(const uint4*)(row + 16);
  return f.v;
}
// A fragment from an f32 global matrix (converted on the fly).
__device__ __forceinline__ v16bf load_a_f32(const float* Xf, int ld, int b0, int kb) {
  const int lane = threadIdx.x & 31;
  const int m = lane & 15, hf = lane >> 4;
  FragB f;
  const float* row = Xf + (size_t)(b0 + m) * ld + kb + 8 * hf;
  #pragma unroll
  for (int i = 0; i < 8; ++i) f.s[i]     = f2bf(row[i]);
  #pragma unroll
  for (int i = 0; i < 8; ++i) f.s[8 + i] = f2bf(row[16 + i]);
  return f.v;
}
// B fragment: 32x16 tile of W^T, i.e. rows n0..n0+15 of row-major bf16 W[N x K].
// Lane covers column n = lane&15; K = 8*half + 0..7 and 16 + 8*half + 0..7 -> two 128-bit loads.
__device__ __forceinline__ v16bf load_b_gbl(const unsigned short* W, int ld, int n0, int kb) {
  const int lane = threadIdx.x & 31;
  const int n = lane & 15, hf = lane >> 4;
  FragB f;
  const unsigned short* col = W + (size_t)(n0 + n) * ld + kb + 8 * hf;
  f.q[0] = *(const uint4*)(col);
  f.q[1] = *(const uint4*)(col + 16);
  return f.v;
}

// Shared head: [mean | raw_std] = sh @ W2^T + b2 (N = 2Z), sample state.
__device__ __forceinline__ void head_sample(
    const unsigned short* s_sh, int LS,
    const unsigned short* W2, const float* b2, const float* nzs,
    float* o_state, float* o_mean, float* o_std, float* st_out,
    int b0, int t) {
  const int wave = threadIdx.x >> 5, lane = threadIdx.x & 31;
  const int nl = lane & 15, hf = lane >> 4;
  #pragma unroll 1
  for (int ti = 0; ti < 2; ++ti) {
    const int n0 = (wave * 2 + ti) * 16;
    v8f cm, cr;
    { float vm = b2[n0 + nl], vr = b2[Z + n0 + nl];
      #pragma unroll
      for (int r = 0; r < 8; ++r) { cm[r] = vm; cr[r] = vr; } }
    for (int kb = 0; kb < H; kb += 32) {
      v16bf a = load_a_lds(s_sh, LS, kb);
      cm = wmma_bf16(a, load_b_gbl(W2, H,     n0, kb), cm);
      cr = wmma_bf16(a, load_b_gbl(W2, H, Z + n0, kb), cr);
    }
    #pragma unroll
    for (int r = 0; r < 8; ++r) {
      int b = b0 + r + 8 * hf;
      float mean = cm[r];
      float sd   = softp(cr[r]) + 0.1f;
      size_t gi  = ((size_t)t * B + b) * Z + n0 + nl;
      float v    = mean + sd * nzs[gi];
      o_state[gi] = v; o_mean[gi] = mean; o_std[gi] = sd;
      if (st_out) st_out[(size_t)b * Z + n0 + nl] = v;
    }
  }
}

// ---------------- prep kernels ----------------
__global__ __launch_bounds__(256) void k_f32_to_bf16(const float* __restrict__ s,
                                                     unsigned short* __restrict__ d, int n) {
  int i = blockIdx.x * 256 + threadIdx.x;
  if (i < n) d[i] = f2bf(s[i]);
}
__global__ __launch_bounds__(256) void k_init(const float* __restrict__ ps,
                                              const float* __restrict__ pb,
                                              float* __restrict__ st32, float* __restrict__ h32,
                                              unsigned short* __restrict__ nb_bf) {
  int i = blockIdx.x * 256 + threadIdx.x;
  if (i < B * Z) st32[i] = ps[i];
  if (i < B * D) { float v = pb[i]; h32[i] = v; nb_bf[i] = f2bf(v); }
}

// ---------------- fully fused RSSM step ----------------
// One workgroup owns 16 batch rows for the whole step: embed -> GRU -> belief
// sample -> prior head -> posterior head.  All carried state is row-local, so
// only workgroup-scope fences are needed around the bf16 global round-trips.
__global__ __launch_bounds__(256) void k_rssm_step(
    const float* __restrict__ st, const float* __restrict__ actions,
    const float* __restrict__ nonterm, const float* __restrict__ obs,
    const float* __restrict__ h32, float* __restrict__ h32_out,
    const unsigned short* __restrict__ hbf, unsigned short* __restrict__ nb_bf,
    unsigned short* __restrict__ rnn_bf, float* __restrict__ st32_out,
    const unsigned short* __restrict__ Wsa, const float* __restrict__ bsa,
    const unsigned short* __restrict__ Wih, const float* __restrict__ bih,
    const unsigned short* __restrict__ Whh, const float* __restrict__ bhh,
    const unsigned short* __restrict__ Web, const float* __restrict__ beb,
    const unsigned short* __restrict__ Wbq, const float* __restrict__ bbq,
    const unsigned short* __restrict__ Webp, const float* __restrict__ bebp,
    const unsigned short* __restrict__ Wsp, const float* __restrict__ bsp,
    const unsigned short* __restrict__ Webq, const float* __restrict__ bebq,
    const unsigned short* __restrict__ Wsq, const float* __restrict__ bsq,
    const float* __restrict__ nzb, const float* __restrict__ nzp,
    const float* __restrict__ nzq,
    float* __restrict__ o_belief,
    float* __restrict__ o_pstate, float* __restrict__ o_pmean, float* __restrict__ o_pstd,
    float* __restrict__ o_qstate, float* __restrict__ o_qmean, float* __restrict__ o_qstd,
    int t) {
  constexpr int LXA = KSA + 8, LX = D + 8, LS = H + 8;
  __shared__ __align__(16) unsigned short s_xa[16 * LXA]; // [s*nt, a] bf16   (9.4 KB)
  __shared__ __align__(16) unsigned short s_x [16 * LX];  // relu embed      (33 KB)
  __shared__ __align__(16) unsigned short s_sh[16 * LS];  // hidden, reused  (16.6 KB)
  const int b0 = blockIdx.x * 16;
  const int wave = threadIdx.x >> 5, lane = threadIdx.x & 31;
  const int nl = lane & 15, hf = lane >> 4;

  // ---- P0: stage [state*nt, action] and prefetch the HBM-resident streams
  // (obs + noise_belief for this workgroup's rows) so they land in cache by P4/P6.
  {
    const float* op = obs + ((size_t)t * B + b0) * E;
    const float* np = nzb + ((size_t)t * B + b0) * D;
    for (int i = threadIdx.x * 32; i < 16 * E; i += 256 * 32) __builtin_prefetch(op + i, 0, 1);
    for (int i = threadIdx.x * 32; i < 16 * D; i += 256 * 32) __builtin_prefetch(np + i, 0, 1);
  }
  for (int idx = threadIdx.x; idx < 16 * KSA; idx += 256) {
    int m = idx / KSA, k = idx - m * KSA, b = b0 + m;
    float v;
    if (k < Z) {
      float nt = (t == 0) ? 1.f : nonterm[(size_t)(t - 1) * B + b];
      v = st[(size_t)b * Z + k] * nt;
    } else {
      v = actions[((size_t)t * B + b) * ACT + (k - Z)];
    }
    s_xa[m * LXA + k] = f2bf(v);
  }
  __syncthreads();

  // ---- P1: x = relu([s,a] @ Wsa^T + bsa) -> LDS
  #pragma unroll 1
  for (int ti = 0; ti < 8; ++ti) {
    const int n0 = (wave * 8 + ti) * 16;
    v8f c; const float bias = bsa[n0 + nl];
    #pragma unroll
    for (int r = 0; r < 8; ++r) c[r] = bias;
    for (int kb = 0; kb < KSA; kb += 32)
      c = wmma_bf16(load_a_lds(s_xa, LXA, kb), load_b_gbl(Wsa, KSA, n0, kb), c);
    #pragma unroll
    for (int r = 0; r < 8; ++r)
      s_x[(r + 8 * hf) * LX + n0 + nl] = f2bf(fmaxf(c[r], 0.f));
  }
  __syncthreads();

  // ---- P2: GRU (six accumulators per tile; gate math in registers) -> rnn_bf
  #pragma unroll 1
  for (int ti = 0; ti < 8; ++ti) {
    const int n0 = (wave * 8 + ti) * 16;
    v8f cir, ciz, cin, dhr, dhz, dhn;
    {
      float v0 = bih[n0 + nl], v1 = bih[D + n0 + nl], v2 = bih[2 * D + n0 + nl];
      float v3 = bhh[n0 + nl], v4 = bhh[D + n0 + nl], v5 = bhh[2 * D + n0 + nl];
      #pragma unroll
      for (int r = 0; r < 8; ++r) { cir[r]=v0; ciz[r]=v1; cin[r]=v2; dhr[r]=v3; dhz[r]=v4; dhn[r]=v5; }
    }
    for (int kb = 0; kb < D; kb += 32) {
      v16bf ax = load_a_lds(s_x, LX, kb);
      v16bf ah = load_a_gbl(hbf, D, b0, kb);
      cir = wmma_bf16(ax, load_b_gbl(Wih, D,         n0, kb), cir);
      ciz = wmma_bf16(ax, load_b_gbl(Wih, D, D     + n0, kb), ciz);
      cin = wmma_bf16(ax, load_b_gbl(Wih, D, 2 * D + n0, kb), cin);
      dhr = wmma_bf16(ah, load_b_gbl(Whh, D,         n0, kb), dhr);
      dhz = wmma_bf16(ah, load_b_gbl(Whh, D, D     + n0, kb), dhz);
      dhn = wmma_bf16(ah, load_b_gbl(Whh, D, 2 * D + n0, kb), dhn);
    }
    #pragma unroll
    for (int r = 0; r < 8; ++r) {
      int b = b0 + r + 8 * hf;
      float hp = h32[(size_t)b * D + n0 + nl];
      float rg = sigm(cir[r] + dhr[r]);
      float zg = sigm(ciz[r] + dhz[r]);
      float ng = tanhf(cin[r] + rg * dhn[r]);
      rnn_bf[(size_t)b * D + n0 + nl] = f2bf((1.f - zg) * ng + zg * hp);
    }
  }
  __threadfence_block();
  __syncthreads();

  // ---- P3: sh = relu(rnn @ Web^T + beb) -> LDS
  #pragma unroll 1
  for (int ti = 0; ti < 4; ++ti) {
    const int n0 = (wave * 4 + ti) * 16;
    v8f c; const float bias = beb[n0 + nl];
    #pragma unroll
    for (int r = 0; r < 8; ++r) c[r] = bias;
    for (int kb = 0; kb < D; kb += 32)
      c = wmma_bf16(load_a_gbl(rnn_bf, D, b0, kb), load_b_gbl(Web, D, n0, kb), c);
    #pragma unroll
    for (int r = 0; r < 8; ++r)
      s_sh[(r + 8 * hf) * LS + n0 + nl] = f2bf(fmaxf(c[r], 0.f));
  }
  __syncthreads();

  // ---- P4: belief sample: new_belief = mean + (softplus(raw)+0.1)*noise
  #pragma unroll 1
  for (int ti = 0; ti < 8; ++ti) {
    const int n0 = (wave * 8 + ti) * 16;
    v8f cm, cr;
    { float vm = bbq[n0 + nl], vr = bbq[D + n0 + nl];
      #pragma unroll
      for (int r = 0; r < 8; ++r) { cm[r] = vm; cr[r] = vr; } }
    for (int kb = 0; kb < H; kb += 32) {
      v16bf a = load_a_lds(s_sh, LS, kb);
      cm = wmma_bf16(a, load_b_gbl(Wbq, H,     n0, kb), cm);
      cr = wmma_bf16(a, load_b_gbl(Wbq, H, D + n0, kb), cr);
    }
    #pragma unroll
    for (int r = 0; r < 8; ++r) {
      int b = b0 + r + 8 * hf;
      float sd = softp(cr[r]) + 0.1f;
      float v = cm[r] + sd * nzb[((size_t)t * B + b) * D + n0 + nl];
      o_belief[((size_t)t * B + b) * D + n0 + nl] = v;
      nb_bf[(size_t)b * D + n0 + nl] = f2bf(v);
      h32_out[(size_t)b * D + n0 + nl] = v;
    }
  }
  __threadfence_block();
  __syncthreads();

  // ---- P5: prior: sh_p = relu(nb @ Webp^T + bebp) -> LDS, then head
  #pragma unroll 1
  for (int ti = 0; ti < 4; ++ti) {
    const int n0 = (wave * 4 + ti) * 16;
    v8f c; const float bias = bebp[n0 + nl];
    #pragma unroll
    for (int r = 0; r < 8; ++r) c[r] = bias;
    for (int kb = 0; kb < D; kb += 32)
      c = wmma_bf16(load_a_gbl(nb_bf, D, b0, kb), load_b_gbl(Webp, D, n0, kb), c);
    #pragma unroll
    for (int r = 0; r < 8; ++r)
      s_sh[(r + 8 * hf) * LS + n0 + nl] = f2bf(fmaxf(c[r], 0.f));
  }
  __syncthreads();
  head_sample(s_sh, LS, Wsp, bsp, nzp, o_pstate, o_pmean, o_pstd, nullptr, b0, t);
  __syncthreads();

  // ---- P6: posterior: sh_q = relu([nb | obs] @ Webq^T + bebq) -> LDS, then head
  const float* ot = obs + (size_t)t * B * E;
  #pragma unroll 1
  for (int ti = 0; ti < 4; ++ti) {
    const int n0 = (wave * 4 + ti) * 16;
    v8f c; const float bias = bebq[n0 + nl];
    #pragma unroll
    for (int r = 0; r < 8; ++r) c[r] = bias;
    for (int kb = 0; kb < D + E; kb += 32) {
      v16bf a = (kb < D) ? load_a_gbl(nb_bf, D, b0, kb)
                         : load_a_f32(ot, E, b0, kb - D);
      c = wmma_bf16(a, load_b_gbl(Webq, D + E, n0, kb), c);
    }
    #pragma unroll
    for (int r = 0; r < 8; ++r)
      s_sh[(r + 8 * hf) * LS + n0 + nl] = f2bf(fmaxf(c[r], 0.f));
  }
  __syncthreads();
  head_sample(s_sh, LS, Wsq, bsq, nzq, o_qstate, o_qmean, o_qstd, st32_out, b0, t);
}

// ---------------- host ----------------
extern "C" void kernel_launch(void* const* d_in, const int* in_sizes, int n_in,
                              void* d_out, int out_size, void* d_ws, size_t ws_size,
                              hipStream_t stream) {
  (void)in_sizes; (void)n_in; (void)out_size; (void)ws_size;
  const float* prev_state   = (const float*)d_in[0];
  const float* prev_belief  = (const float*)d_in[1];
  const float* actions      = (const float*)d_in[2];
  const float* observations = (const float*)d_in[3];
  const float* nonterminals = (const float*)d_in[4];
  const float* noise_belief = (const float*)d_in[5];
  const float* noise_prior  = (const float*)d_in[6];
  const float* noise_post   = (const float*)d_in[7];
  const float* W_sa = (const float*)d_in[8];   const float* b_sa = (const float*)d_in[9];
  const float* W_ih = (const float*)d_in[10];  const float* b_ih = (const float*)d_in[11];
  const float* W_hh = (const float*)d_in[12];  const float* b_hh = (const float*)d_in[13];
  const float* W_eb = (const float*)d_in[14];  const float* b_eb = (const float*)d_in[15];
  const float* W_bq = (const float*)d_in[16];  const float* b_bq = (const float*)d_in[17];
  const float* W_ebp= (const float*)d_in[18];  const float* b_ebp= (const float*)d_in[19];
  const float* W_sp = (const float*)d_in[20];  const float* b_sp = (const float*)d_in[21];
  const float* W_ebq= (const float*)d_in[22];  const float* b_ebq= (const float*)d_in[23];
  const float* W_sq = (const float*)d_in[24];  const float* b_sq = (const float*)d_in[25];

  char* ws = (char*)d_ws; size_t off = 0;
  auto carve = [&](size_t bytes) { char* p = ws + off; off += (bytes + 255) & ~(size_t)255; return p; };
  unsigned short* wsa  = (unsigned short*)carve((size_t)D * KSA * 2);
  unsigned short* wih  = (unsigned short*)carve((size_t)3 * D * D * 2);
  unsigned short* whh  = (unsigned short*)carve((size_t)3 * D * D * 2);
  unsigned short* web  = (unsigned short*)carve((size_t)H * D * 2);
  unsigned short* wbq  = (unsigned short*)carve((size_t)2 * D * H * 2);
  unsigned short* webp = (unsigned short*)carve((size_t)H * D * 2);
  unsigned short* wsp  = (unsigned short*)carve((size_t)2 * Z * H * 2);
  unsigned short* webq = (unsigned short*)carve((size_t)H * (D + E) * 2);
  unsigned short* wsq  = (unsigned short*)carve((size_t)2 * Z * H * 2);
  unsigned short* rnn_bf = (unsigned short*)carve((size_t)B * D * 2);
  unsigned short* nb_bf  = (unsigned short*)carve((size_t)B * D * 2);
  float* h32  = (float*)carve((size_t)B * D * 4);
  float* st32 = (float*)carve((size_t)B * Z * 4);

  struct CV { const float* s; unsigned short* d; int n; } cv[9] = {
    {W_sa, wsa, D * KSA}, {W_ih, wih, 3 * D * D}, {W_hh, whh, 3 * D * D},
    {W_eb, web, H * D},   {W_bq, wbq, 2 * D * H}, {W_ebp, webp, H * D},
    {W_sp, wsp, 2 * Z * H}, {W_ebq, webq, H * (D + E)}, {W_sq, wsq, 2 * Z * H}};
  for (int i = 0; i < 9; ++i)
    k_f32_to_bf16<<<(cv[i].n + 255) / 256, 256, 0, stream>>>(cv[i].s, cv[i].d, cv[i].n);
  k_init<<<(B * D + 255) / 256, 256, 0, stream>>>(prev_state, prev_belief, st32, h32, nb_bf);

  float* out      = (float*)d_out;
  float* o_belief = out;
  float* o_pstate = o_belief + (size_t)T * B * D;
  float* o_pmean  = o_pstate + (size_t)T * B * Z;
  float* o_pstd   = o_pmean  + (size_t)T * B * Z;
  float* o_qstate = o_pstd   + (size_t)T * B * Z;
  float* o_qmean  = o_qstate + (size_t)T * B * Z;
  float* o_qstd   = o_qmean  + (size_t)T * B * Z;

  for (int t = 0; t < T; ++t) {
    // h / new_belief live in the same buffers across steps: a step reads the
    // previous step's values (P2) strictly before overwriting them (P4), and
    // the rows involved are owned by a single workgroup.
    k_rssm_step<<<B / 16, 256, 0, stream>>>(
        st32, actions, nonterminals, observations,
        h32, h32, nb_bf, nb_bf, rnn_bf, st32,
        wsa, b_sa, wih, b_ih, whh, b_hh,
        web, b_eb, wbq, b_bq,
        webp, b_ebp, wsp, b_sp,
        webq, b_ebq, wsq, b_sq,
        noise_belief, noise_prior, noise_post,
        o_belief, o_pstate, o_pmean, o_pstd,
        o_qstate, o_qmean, o_qstd, t);
  }
}
